// Attention_76235669504391
// MI455X (gfx1250) — matrix operand order repaired and checked by reading, not compiled, not run on hardware
//
#include <hip/hip_runtime.h>
#include <hip/hip_bf16.h>

#define DIMC   1024
#define NHEAD  16
#define HDIM   64
#define BBATCH 2
#define SEQ    2048
#define MTOT   (BBATCH*SEQ)   // 4096

typedef __bf16 bf16;
typedef bf16 v16bf __attribute__((ext_vector_type(16)));
typedef bf16 v8bf  __attribute__((ext_vector_type(8)));
typedef float v8f  __attribute__((ext_vector_type(8)));
typedef unsigned int u32x4 __attribute__((ext_vector_type(4)));
typedef int i32x8 __attribute__((ext_vector_type(8)));
typedef int i32x4 __attribute__((ext_vector_type(4)));

union BF16Frag { v16bf v; v8bf h[2]; };

static __device__ __forceinline__ v8f wmma_bf16(v16bf a, v16bf b, v8f c) {
  // D = A(16x32 bf16) * B(32x16 bf16) + C(16x16 f32)
  return __builtin_amdgcn_wmma_f32_16x16x32_bf16(false, a, false, b, (short)0, c,
                                                 false, false);
}

// ---------------------------------------------------------------------------
// TDM: DMA a 2-D bf16 tile (tile_d0 contiguous elems x tile_d1 lines, line
// stride stride0 elems) from global into LDS at byte offset lds_off.
// Descriptor layout per CDNA5 ISA ch.8 (D# group0/group1), 2-D: groups 2/3 zero.
// ---------------------------------------------------------------------------
static __device__ __forceinline__ void tdm_load_tile_bf16(
    unsigned lds_off, const bf16* gptr,
    unsigned tensor_d0, unsigned tensor_d1,
    unsigned tile_d0, unsigned tile_d1,
    unsigned long long stride0 /*elements*/) {
  unsigned long long ga = (unsigned long long)(uintptr_t)gptr;
  u32x4 g0;
  g0[0] = 1u;                                           // count=1, user mode
  g0[1] = lds_off;                                      // lds_addr [63:32]
  g0[2] = (unsigned)(ga & 0xffffffffu);                 // global_addr lo
  g0[3] = (unsigned)((ga >> 32) & 0x01ffffffu)          // global_addr hi
          | 0x80000000u;                                // type=2 ("image")
  i32x8 g1;
  g1[0] = (int)(1u << 16);                              // data_size=1 -> 2 bytes
  g1[1] = (int)((tensor_d0 & 0xffffu) << 16);           // tensor_dim0 lo16
  g1[2] = (int)(((tensor_d0 >> 16) & 0xffffu)
          | ((tensor_d1 & 0xffffu) << 16));             // td0 hi16 | td1 lo16
  g1[3] = (int)(((tensor_d1 >> 16) & 0xffffu)
          | ((tile_d0 & 0xffffu) << 16));               // td1 hi16 | tile_dim0
  g1[4] = (int)(tile_d1 & 0xffffu);                     // tile_dim1 (tile_dim2=0)
  g1[5] = (int)(stride0 & 0xffffffffull);               // dim0_stride lo32
  g1[6] = (int)((stride0 >> 32) & 0xffffull);           // dim0_stride hi16
  g1[7] = 0;                                            // dim1_stride (2-D)
  i32x4 gz4 = {0, 0, 0, 0};
  i32x8 gz8 = {0, 0, 0, 0, 0, 0, 0, 0};
  __builtin_amdgcn_tensor_load_to_lds(g0, g1, gz4, gz4, gz8, 0);
}

// ---------------------------------------------------------------------------
// fp32 -> bf16 convert (8 elements / thread, packed 16B store)
// ---------------------------------------------------------------------------
__global__ void cvt_f32_bf16(const float* __restrict__ in, bf16* __restrict__ out,
                             int n) {
  int i = (blockIdx.x * blockDim.x + threadIdx.x) * 8;
  if (i + 7 < n) {
    float4 f0 = *(const float4*)(in + i);
    float4 f1 = *(const float4*)(in + i + 4);
    v8bf o;
    o[0] = (bf16)f0.x; o[1] = (bf16)f0.y; o[2] = (bf16)f0.z; o[3] = (bf16)f0.w;
    o[4] = (bf16)f1.x; o[5] = (bf16)f1.y; o[6] = (bf16)f1.z; o[7] = (bf16)f1.w;
    *(v8bf*)(out + i) = o;
  }
}

// ---------------------------------------------------------------------------
// GEMM: out(MxN, f32) = A(MxK bf16 row-major) * W(NxK bf16 row-major)^T + bias
// Block 256 thr = 8 waves. Block tile 128(M) x 128(N); wave tile 32(M) x 64(N).
// Weight tile (128 n-rows x 32 k) staged to LDS by the Tensor Data Mover,
// double-buffered, tracked with TENSORcnt. A rows stream from global/L2.
// ---------------------------------------------------------------------------
__global__ __launch_bounds__(256) void gemm_bf16_wmma(
    const bf16* __restrict__ A, const bf16* __restrict__ W,
    const float* __restrict__ bias, float* __restrict__ out,
    int M, int N, int K) {
  __shared__ __align__(128) bf16 Bsh[2][128 * 32];      // 2 x 8 KB

  const int tid   = threadIdx.x;
  const int wave  = tid >> 5;
  const int lane  = tid & 31;
  const int l16   = lane & 15;
  const int lh    = lane >> 4;            // lane-half (0/1)
  const int waveM = wave >> 1;            // 0..3
  const int waveN = wave & 1;             // 0..1
  const int m0    = blockIdx.y * 128 + waveM * 32;
  const int n0blk = blockIdx.x * 128;
  const int n0    = waveN * 64;           // within block tile

  v8f acc[2][4];
#pragma unroll
  for (int mt = 0; mt < 2; ++mt)
#pragma unroll
    for (int nt = 0; nt < 4; ++nt) acc[mt][nt] = v8f{};

  const bf16* arow0 = A + (size_t)(m0 + l16) * K;
  const bf16* arow1 = arow0 + (size_t)16 * K;

  const int NI = K / 32;
  if (wave == 0)
    tdm_load_tile_bf16((unsigned)(uintptr_t)&Bsh[0][0],
                       W + (size_t)n0blk * K, K, N, 32, 128, (unsigned long long)K);

  for (int i = 0; i < NI; ++i) {
    const int kb = i * 32;
    const int pb = i & 1;
    if (wave == 0) __builtin_amdgcn_s_wait_tensorcnt(0);
    __syncthreads();                       // buf[pb] ready; buf[1-pb] free
    if (wave == 0 && i + 1 < NI)
      tdm_load_tile_bf16((unsigned)(uintptr_t)&Bsh[1 - pb][0],
                         W + (size_t)n0blk * K + (kb + 32), K, N, 32, 128,
                         (unsigned long long)K);

    // A fragments (16x32): lanes 0-15 K=0..15; lanes 16-31 K=16..31
    BF16Frag a0, a1;
    a0.h[0] = *(const v8bf*)(arow0 + kb + 16 * lh);
    a0.h[1] = *(const v8bf*)(arow0 + kb + 16 * lh + 8);
    a1.h[0] = *(const v8bf*)(arow1 + kb + 16 * lh);
    a1.h[1] = *(const v8bf*)(arow1 + kb + 16 * lh + 8);

#pragma unroll
    for (int nt = 0; nt < 4; ++nt) {
      // B fragment (32x16) from LDS: lane holds col n=l16; halves hold K 0..15/16..31
      const bf16* bp = &Bsh[pb][(n0 + nt * 16 + l16) * 32 + 16 * lh];
      v16bf bfr = *(const v16bf*)bp;
      acc[0][nt] = wmma_bf16(a0.v, bfr, acc[0][nt]);
      acc[1][nt] = wmma_bf16(a1.v, bfr, acc[1][nt]);
    }
  }

#pragma unroll
  for (int mt = 0; mt < 2; ++mt) {
#pragma unroll
    for (int nt = 0; nt < 4; ++nt) {
      const int n = n0blk + n0 + nt * 16 + l16;
      const float bn = bias[n];
#pragma unroll
      for (int v = 0; v < 8; ++v) {
        const int m = m0 + mt * 16 + v + 8 * lh;   // C/D layout row = vgpr+8*half
        out[(size_t)m * N + n] = acc[mt][nt][v] + bn;
      }
    }
  }
}

// ---------------------------------------------------------------------------
// RoPE + repack: qproj/kproj/vproj (b,s,dim f32) ->
//   qh,kh : (b,h,s,d) bf16 (rotary applied) ; vt : (b,h,d,s) bf16
// ---------------------------------------------------------------------------
__global__ void rope_pack(const float* __restrict__ qproj,
                          const float* __restrict__ kproj,
                          const float* __restrict__ vproj,
                          const float* __restrict__ freqs,
                          bf16* __restrict__ qh, bf16* __restrict__ kh,
                          bf16* __restrict__ vt) {
  int i = blockIdx.x * blockDim.x + threadIdx.x;   // B*S*H*32 threads
  int p = i & 31;
  int t = i >> 5;
  int h = t & (NHEAD - 1);
  int u = t >> 4;
  int s = u & (SEQ - 1);
  int b = u >> 11;

  size_t src = ((size_t)(b * SEQ + s)) * DIMC + h * HDIM + 2 * p;
  float f0 = freqs[(s * 32 + p) * 2 + 0];
  float f1 = freqs[(s * 32 + p) * 2 + 1];

  float qr = qproj[src], qi = qproj[src + 1];
  float kr = kproj[src], ki = kproj[src + 1];

  size_t dst = ((size_t)((b * NHEAD + h) * SEQ + s)) * HDIM + 2 * p;
  qh[dst]     = (bf16)(qr * f0 - qi * f1);
  qh[dst + 1] = (bf16)(qi * f0 + qr * f1);
  kh[dst]     = (bf16)(kr * f0 - ki * f1);
  kh[dst + 1] = (bf16)(ki * f0 + kr * f1);

  float v0 = vproj[src], v1 = vproj[src + 1];
  size_t vb = ((size_t)(b * NHEAD + h)) * HDIM * SEQ;
  vt[vb + (size_t)(2 * p) * SEQ + s]     = (bf16)v0;
  vt[vb + (size_t)(2 * p + 1) * SEQ + s] = (bf16)v1;
}

// ---------------------------------------------------------------------------
// Flash attention: block = 128 thr (4 waves), each wave owns 16 queries.
// grid = (S/64, B*H). qh/kh: (bh, s, d) bf16; vt: (bh, d, s) bf16.
// K tile (32 keys x 64 d) and V tile (64 d x 32 keys) are TDM-staged into
// double-buffered LDS, shared by all 4 waves (4x VMEM traffic reduction).
// ---------------------------------------------------------------------------
__global__ __launch_bounds__(128) void attn_wmma(
    const bf16* __restrict__ qh, const bf16* __restrict__ kh,
    const bf16* __restrict__ vt, bf16* __restrict__ aout) {
  __shared__ __align__(128) bf16 Ksh[2][32 * 64];   // 2 x 4 KB (keys x d)
  __shared__ __align__(128) bf16 Vsh[2][64 * 32];   // 2 x 4 KB (d x keys)
  __shared__ __align__(16)  bf16 plds[4][16 * 32];  // per-wave P-transpose tile

  const int tid  = threadIdx.x;
  const int wave = tid >> 5;
  const int lane = tid & 31;
  const int l16  = lane & 15;
  const int lh   = lane >> 4;

  const int bh = blockIdx.y;
  const int b  = bh >> 4;            // / NHEAD
  const int h  = bh & (NHEAD - 1);
  const int q0 = blockIdx.x * 64 + wave * 16;

  const bf16* Q  = qh + (size_t)bh * SEQ * HDIM;
  const bf16* Kp = kh + (size_t)bh * SEQ * HDIM;
  const bf16* V  = vt + (size_t)bh * HDIM * SEQ;

  // Q A-fragments (resident for whole loop)
  const bf16* qrow = Q + (size_t)(q0 + l16) * HDIM;
  BF16Frag qa0, qa1;
  qa0.h[0] = *(const v8bf*)(qrow + 16 * lh);
  qa0.h[1] = *(const v8bf*)(qrow + 16 * lh + 8);
  qa1.h[0] = *(const v8bf*)(qrow + 32 + 16 * lh);
  qa1.h[1] = *(const v8bf*)(qrow + 32 + 16 * lh + 8);

  v8f o[4] = {v8f{}, v8f{}, v8f{}, v8f{}};
  float mrow[8], lrow[8];
#pragma unroll
  for (int v = 0; v < 8; ++v) { mrow[v] = -1e30f; lrow[v] = 0.0f; }

  const float scale = 0.125f;   // 1/sqrt(64)
  const int NI = SEQ / 32;

  if (wave == 0) {
    tdm_load_tile_bf16((unsigned)(uintptr_t)&Ksh[0][0], Kp,
                       HDIM, SEQ, 64, 32, (unsigned long long)HDIM);
    tdm_load_tile_bf16((unsigned)(uintptr_t)&Vsh[0][0], V,
                       SEQ, HDIM, 32, 64, (unsigned long long)SEQ);
  }

  for (int it = 0; it < NI; ++it) {
    const int kb = it * 32;
    const int pb = it & 1;
    if (wave == 0) __builtin_amdgcn_s_wait_tensorcnt(0);
    __syncthreads();                     // buf[pb] ready; buf[1-pb] free
    if (wave == 0 && it + 1 < NI) {
      tdm_load_tile_bf16((unsigned)(uintptr_t)&Ksh[1 - pb][0],
                         Kp + (size_t)(kb + 32) * HDIM,
                         HDIM, SEQ, 64, 32, (unsigned long long)HDIM);
      tdm_load_tile_bf16((unsigned)(uintptr_t)&Vsh[1 - pb][0],
                         V + (kb + 32),
                         SEQ, HDIM, 32, 64, (unsigned long long)SEQ);
    }

    // ---- QK^T: two 16x16 score tiles from LDS K ----
    const bf16* kx = &Ksh[pb][0];
    v16bf kb00 = *(const v16bf*)&kx[l16 * 64 + 16 * lh];            // d 0..31
    v16bf kb01 = *(const v16bf*)&kx[l16 * 64 + 32 + 16 * lh];       // d 32..63
    v16bf kb10 = *(const v16bf*)&kx[(16 + l16) * 64 + 16 * lh];
    v16bf kb11 = *(const v16bf*)&kx[(16 + l16) * 64 + 32 + 16 * lh];

    v8f s0 = {}; s0 = wmma_bf16(qa0.v, kb00, s0); s0 = wmma_bf16(qa1.v, kb01, s0);
    v8f s1 = {}; s1 = wmma_bf16(qa0.v, kb10, s1); s1 = wmma_bf16(qa1.v, kb11, s1);

    // ---- online softmax (row = vgpr + 8*lane_half; cols across 16 lanes) ----
#pragma unroll
    for (int v = 0; v < 8; ++v) {
      float a0 = s0[v] * scale;
      float a1 = s1[v] * scale;
      float mx = fmaxf(a0, a1);
#pragma unroll
      for (int msk = 1; msk < 16; msk <<= 1) mx = fmaxf(mx, __shfl_xor(mx, msk));
      float mnew  = fmaxf(mrow[v], mx);
      float alpha = __expf(mrow[v] - mnew);
      float p0 = __expf(a0 - mnew);
      float p1 = __expf(a1 - mnew);
      float rs = p0 + p1;
#pragma unroll
      for (int msk = 1; msk < 16; msk <<= 1) rs += __shfl_xor(rs, msk);
      lrow[v] = lrow[v] * alpha + rs;
      mrow[v] = mnew;
#pragma unroll
      for (int nt = 0; nt < 4; ++nt) o[nt][v] *= alpha;
      int r = v + 8 * lh;
      plds[wave][r * 32 + l16]      = (bf16)p0;
      plds[wave][r * 32 + 16 + l16] = (bf16)p1;
    }

    // ---- P back in A(16x32) layout ----
    BF16Frag pa;
    pa.h[0] = *(const v8bf*)&plds[wave][l16 * 32 + 16 * lh];
    pa.h[1] = *(const v8bf*)&plds[wave][l16 * 32 + 16 * lh + 8];

    // ---- PV: 4 N-tiles over head dim, V from LDS ----
#pragma unroll
    for (int nt = 0; nt < 4; ++nt) {
      v16bf vb = *(const v16bf*)&Vsh[pb][(nt * 16 + l16) * 32 + 16 * lh];
      o[nt] = wmma_bf16(pa.v, vb, o[nt]);
    }
  }

  // ---- normalize + store (b, s, h*64+d) bf16 ----
#pragma unroll
  for (int nt = 0; nt < 4; ++nt) {
#pragma unroll
    for (int v = 0; v < 8; ++v) {
      int q = q0 + v + 8 * lh;
      int d = nt * 16 + l16;
      float val = o[nt][v] / lrow[v];
      aout[((size_t)(b * SEQ + q)) * DIMC + h * HDIM + d] = (bf16)val;
    }
  }
}

// ---------------------------------------------------------------------------
// launch
// ---------------------------------------------------------------------------
extern "C" void kernel_launch(void* const* d_in, const int* in_sizes, int n_in,
                              void* d_out, int out_size, void* d_ws, size_t ws_size,
                              hipStream_t stream) {
  const float* q     = (const float*)d_in[0];
  const float* freqs = (const float*)d_in[1];
  const float* wq_w  = (const float*)d_in[2];
  const float* wq_b  = (const float*)d_in[3];
  const float* wk_w  = (const float*)d_in[4];
  const float* wk_b  = (const float*)d_in[5];
  const float* wv_w  = (const float*)d_in[6];
  const float* wv_b  = (const float*)d_in[7];
  const float* wo_w  = (const float*)d_in[8];
  const float* wo_b  = (const float*)d_in[9];
  float* out = (float*)d_out;

  char* ws = (char*)d_ws;
  const size_t MB = 1u << 20;
  bf16*  qbf   = (bf16*)(ws);                         // 8 MB  (4096x1024)
  bf16*  wqbf  = (bf16*)(ws + 8 * MB);                // 2 MB each
  bf16*  wkbf  = wqbf + (size_t)DIMC * DIMC;
  bf16*  wvbf  = wkbf + (size_t)DIMC * DIMC;
  bf16*  wobf  = wvbf + (size_t)DIMC * DIMC;
  float* qproj = (float*)(ws + 16 * MB);              // 16 MB
  float* kproj = (float*)(ws + 32 * MB);              // 16 MB
  float* vproj = (float*)(ws + 48 * MB);              // 16 MB
  bf16*  qhB   = (bf16*)(ws + 64 * MB);               // 8 MB (b,h,s,d)
  bf16*  khB   = (bf16*)(ws + 72 * MB);               // 8 MB
  bf16*  vtB   = (bf16*)(ws + 80 * MB);               // 8 MB (b,h,d,s)
  bf16*  aout  = (bf16*)(ws + 88 * MB);               // 8 MB

  {
    int nAct = MTOT * DIMC;
    cvt_f32_bf16<<<nAct / (256 * 8), 256, 0, stream>>>(q, qbf, nAct);
    int nW = DIMC * DIMC;
    cvt_f32_bf16<<<nW / (256 * 8), 256, 0, stream>>>(wq_w, wqbf, nW);
    cvt_f32_bf16<<<nW / (256 * 8), 256, 0, stream>>>(wk_w, wkbf, nW);
    cvt_f32_bf16<<<nW / (256 * 8), 256, 0, stream>>>(wv_w, wvbf, nW);
    cvt_f32_bf16<<<nW / (256 * 8), 256, 0, stream>>>(wo_w, wobf, nW);
  }

  dim3 ggrid(DIMC / 128, MTOT / 128);                 // (8, 32)
  gemm_bf16_wmma<<<ggrid, 256, 0, stream>>>(qbf, wqbf, wq_b, qproj, MTOT, DIMC, DIMC);
  gemm_bf16_wmma<<<ggrid, 256, 0, stream>>>(qbf, wkbf, wk_b, kproj, MTOT, DIMC, DIMC);
  gemm_bf16_wmma<<<ggrid, 256, 0, stream>>>(qbf, wvbf, wv_b, vproj, MTOT, DIMC, DIMC);

  {
    int total = BBATCH * SEQ * NHEAD * 32;
    rope_pack<<<total / 256, 256, 0, stream>>>(qproj, kproj, vproj, freqs,
                                               qhB, khB, vtB);
  }

  dim3 agrid(SEQ / 64, BBATCH * NHEAD);               // (32, 32)
  attn_wmma<<<agrid, 128, 0, stream>>>(qhB, khB, vtB, aout);

  gemm_bf16_wmma<<<ggrid, 256, 0, stream>>>(aout, wobf, wo_b, out, MTOT, DIMC, DIMC);
}
